// Attention_2894807958101
// MI455X (gfx1250) — compile-verified, matrix operand order checked by reading, not run on hardware
//
#include <hip/hip_runtime.h>
#include <hip/hip_bf16.h>
#include <cstdint>

#define D_MODEL 1024
#define N_HEADS 16
#define HEAD_DIM 64
#define BATCH 4
#define SEQ 2048
#define LN_EPS 1e-6f

typedef __attribute__((ext_vector_type(16))) __bf16 v16bf;
typedef __attribute__((ext_vector_type(8)))  __bf16 v8bf;
typedef __attribute__((ext_vector_type(8)))  float  v8f;

union Frag16 { v16bf v; v8bf h[2]; };

// Build a 16-half WMMA fragment from two contiguous 16-byte chunks
// (halves 0..7 = K [k0..k0+7], halves 8..15 = K [k0+16..k0+23] per ISA layout).
__device__ __forceinline__ v16bf load_frag(const __bf16* p0, const __bf16* p1) {
    Frag16 f;
    f.h[0] = *(const v8bf*)p0;
    f.h[1] = *(const v8bf*)p1;
    return f.v;
}

// Async 16B global -> LDS copy (per lane).  IOFFSET is added to BOTH the LDS
// and the global address (ISA 15.18.3), so one LDS-offset reg + one global
// base covers a contiguous segment with immediate offsets.
#define ASYNC_CP16(ldsoff, gaddr, imm)                                        \
    asm volatile("global_load_async_to_lds_b128 %0, %1, off offset:%2"        \
                 :: "v"(ldsoff), "v"(gaddr), "i"(imm) : "memory")
#define WAIT_ASYNC0() asm volatile("s_wait_asynccnt 0x0" ::: "memory")
#define WAIT_DS0()    asm volatile("s_wait_dscnt 0x0" ::: "memory")

__device__ __forceinline__ uint32_t lds_off(const void* p) {
    return (uint32_t)(uintptr_t)p;   // low 32 bits of generic LDS addr = offset
}

__device__ __forceinline__ float rowmax16(float v) {
#pragma unroll
    for (int o = 8; o > 0; o >>= 1) v = fmaxf(v, __shfl_xor(v, o, 16));
    return v;
}
__device__ __forceinline__ float rowsum16(float v) {
#pragma unroll
    for (int o = 8; o > 0; o >>= 1) v += __shfl_xor(v, o, 16);
    return v;
}

// ---------------------------------------------------------------------------
// LayerNorm (fp32 in) -> bf16 out.  One block (256 threads) per row of 1024.
// ---------------------------------------------------------------------------
__global__ __launch_bounds__(256)
void ln_bf16_kernel(const float* __restrict__ x,
                    const float* __restrict__ gamma,
                    const float* __restrict__ beta,
                    __bf16* __restrict__ out) {
    __shared__ float rs[8], rq[8];
    const int row  = blockIdx.x;
    const int tid  = threadIdx.x;
    const int lane = tid & 31;
    const int wv   = tid >> 5;

    const float* xr = x + (size_t)row * D_MODEL;
    float4 v = ((const float4*)xr)[tid];

    float s  = v.x + v.y + v.z + v.w;
    float sq = v.x * v.x + v.y * v.y + v.z * v.z + v.w * v.w;
#pragma unroll
    for (int o = 16; o > 0; o >>= 1) {
        s  += __shfl_xor(s, o, 32);
        sq += __shfl_xor(sq, o, 32);
    }
    if (lane == 0) { rs[wv] = s; rq[wv] = sq; }
    __syncthreads();
    float ts = 0.f, tq = 0.f;
#pragma unroll
    for (int i = 0; i < 8; ++i) { ts += rs[i]; tq += rq[i]; }

    const float mean = ts * (1.0f / D_MODEL);
    const float var  = tq * (1.0f / D_MODEL) - mean * mean;
    const float rstd = rsqrtf(var + LN_EPS);

    const int i = tid * 4;
    __bf16* orow = out + (size_t)row * D_MODEL;
    orow[i + 0] = (__bf16)((v.x - mean) * rstd * gamma[i + 0] + beta[i + 0]);
    orow[i + 1] = (__bf16)((v.y - mean) * rstd * gamma[i + 1] + beta[i + 1]);
    orow[i + 2] = (__bf16)((v.z - mean) * rstd * gamma[i + 2] + beta[i + 2]);
    orow[i + 3] = (__bf16)((v.w - mean) * rstd * gamma[i + 3] + beta[i + 3]);
}

// ---------------------------------------------------------------------------
// fp32 -> bf16 conversion (weights).
// ---------------------------------------------------------------------------
__global__ void f32_to_bf16_kernel(const float* __restrict__ src,
                                   __bf16* __restrict__ dst, int n) {
    int i = blockIdx.x * blockDim.x + threadIdx.x;
    const int stride = gridDim.x * blockDim.x;
    for (; i < n; i += stride) dst[i] = (__bf16)src[i];
}

// ---------------------------------------------------------------------------
// Tiled bf16 GEMM:  Y[m,n] = sum_k X[m,k] * W[n,k] + bias[n]
//   Fixed N=K=1024.  Block tile 128x128, 8 waves, wave tile 32x64 (2x4 accs).
//   64-wide K steps, DOUBLE-BUFFERED async global->LDS staging: next tile's
//   copies are issued right after the barrier, fully overlapping global
//   latency with the 16 WMMAs of the current tile.  One barrier per K step.
// ---------------------------------------------------------------------------
#define GK   1024
#define GN   1024
#define ASTR 72

template <bool F32OUT>
__global__ __launch_bounds__(256)
void gemm_bf16_kernel(const __bf16* __restrict__ X,
                      const __bf16* __restrict__ W,
                      const float* __restrict__ bias,
                      void* __restrict__ Yv) {
    __shared__ __bf16 As[2][128 * ASTR];
    __shared__ __bf16 Bs[2][128 * ASTR];

    const int tid  = threadIdx.x;
    const int lane = tid & 31;
    const int wave = tid >> 5;
    const int wm = wave >> 1;          // 0..3  (M direction, 32 rows each)
    const int wn = wave & 1;           // 0..1  (N direction, 64 cols each)
    const int m0 = blockIdx.y * 128;
    const int n0 = blockIdx.x * 128;

    const int mrow = lane & 15;
    const int ksel = lane >> 4;
    const int half = ksel;

    const int lrow = tid >> 1;         // 0..127 staging row
    const int lseg = tid & 1;          // which 32-half segment of the 64 K

    v8f acc[2][4];
#pragma unroll
    for (int mt = 0; mt < 2; ++mt)
#pragma unroll
        for (int nt = 0; nt < 4; ++nt) acc[mt][nt] = {};

    const __bf16* ag = X + (size_t)(m0 + lrow) * GK + lseg * 32;
    const __bf16* bg = W + (size_t)(n0 + lrow) * GK + lseg * 32;
    uint32_t aoff[2], boff[2];
#pragma unroll
    for (int u = 0; u < 2; ++u) {
        aoff[u] = lds_off(&As[u][lrow * ASTR + lseg * 32]);
        boff[u] = lds_off(&Bs[u][lrow * ASTR + lseg * 32]);
    }

    // Prologue: stage tile 0 into buffer 0.
    {
        const uint64_t ga = (uint64_t)(uintptr_t)ag;
        const uint64_t gb = (uint64_t)(uintptr_t)bg;
        ASYNC_CP16(aoff[0], ga, 0);  ASYNC_CP16(aoff[0], ga, 16);
        ASYNC_CP16(aoff[0], ga, 32); ASYNC_CP16(aoff[0], ga, 48);
        ASYNC_CP16(boff[0], gb, 0);  ASYNC_CP16(boff[0], gb, 16);
        ASYNC_CP16(boff[0], gb, 32); ASYNC_CP16(boff[0], gb, 48);
    }

    for (int k0 = 0; k0 < GK; k0 += 64) {
        const int it = (k0 >> 6) & 1;
        WAIT_ASYNC0();
        __syncthreads();

        if (k0 + 64 < GK) {            // prefetch next tile into other buffer
            const int nb = it ^ 1;
            const uint64_t ga = (uint64_t)(uintptr_t)(ag + k0 + 64);
            const uint64_t gb = (uint64_t)(uintptr_t)(bg + k0 + 64);
            ASYNC_CP16(aoff[nb], ga, 0);  ASYNC_CP16(aoff[nb], ga, 16);
            ASYNC_CP16(aoff[nb], ga, 32); ASYNC_CP16(aoff[nb], ga, 48);
            ASYNC_CP16(boff[nb], gb, 0);  ASYNC_CP16(boff[nb], gb, 16);
            ASYNC_CP16(boff[nb], gb, 32); ASYNC_CP16(boff[nb], gb, 48);
        }

#pragma unroll
        for (int kk = 0; kk < 2; ++kk) {
            const int ko = kk * 32 + ksel * 8;
            v16bf af[2], bf[4];
#pragma unroll
            for (int mt = 0; mt < 2; ++mt) {
                const int r = wm * 32 + mt * 16 + mrow;
                af[mt] = load_frag(&As[it][r * ASTR + ko],
                                   &As[it][r * ASTR + ko + 16]);
            }
#pragma unroll
            for (int nt = 0; nt < 4; ++nt) {
                const int c = wn * 64 + nt * 16 + mrow;
                bf[nt] = load_frag(&Bs[it][c * ASTR + ko],
                                   &Bs[it][c * ASTR + ko + 16]);
            }
#pragma unroll
            for (int mt = 0; mt < 2; ++mt)
#pragma unroll
                for (int nt = 0; nt < 4; ++nt)
                    acc[mt][nt] = __builtin_amdgcn_wmma_f32_16x16x32_bf16(
                        false, af[mt], false, bf[nt], (short)0, acc[mt][nt],
                        false, false);
        }
    }

    // Epilogue: one base pointer per (mt,nt), immediate row offsets.
#pragma unroll
    for (int mt = 0; mt < 2; ++mt) {
#pragma unroll
        for (int nt = 0; nt < 4; ++nt) {
            const int col = n0 + wn * 64 + nt * 16 + mrow;
            const float bv = bias[col];
            const size_t base =
                (size_t)(m0 + wm * 32 + mt * 16 + half * 8) * GN + col;
            if (F32OUT) {
                float* p = (float*)Yv + base;
#pragma unroll
                for (int r = 0; r < 8; ++r) p[r * GN] = acc[mt][nt][r] + bv;
            } else {
                __bf16* p = (__bf16*)Yv + base;
#pragma unroll
                for (int r = 0; r < 8; ++r)
                    p[r * GN] = (__bf16)(acc[mt][nt][r] + bv);
            }
        }
    }
}

// ---------------------------------------------------------------------------
// Flash-style attention with sigmoid-gated exp weights:
//   w = sigmoid(s - mu) * exp(s),  out = (w @ V) / rowsum(w),  s = qk/8
// Block = 128 threads (4 waves), 64 query rows per (b,h); keys in 64-steps.
// Double-buffered K (async->LDS) and V (global->VGPR->LDS transpose).
// ---------------------------------------------------------------------------
#define KSTR 72

__global__ __launch_bounds__(128)
void attn_kernel(const __bf16* __restrict__ Q,
                 const __bf16* __restrict__ Kg,
                 const __bf16* __restrict__ V,
                 const float* __restrict__ mu_p,
                 __bf16* __restrict__ O) {
    __shared__ __bf16 Ks[2][64 * KSTR];     // [key][hd]
    __shared__ __bf16 Vt[2][64 * KSTR];     // [hd][key] (transposed)
    __shared__ __bf16 Ps[4][16 * KSTR];     // per-wave P tile [16][64]

    const int tid  = threadIdx.x;
    const int lane = tid & 31;
    const int wv   = tid >> 5;
    const int qb = blockIdx.x & 31;               // SEQ/64 = 32
    const int h  = (blockIdx.x >> 5) & (N_HEADS - 1);
    const int b  = blockIdx.x >> 9;
    const float muv = mu_p[0];

    const int mrow = lane & 15;
    const int ksel = lane >> 4;
    const int half = ksel;

    const size_t rowbase = (size_t)b * SEQ;
    const int q0   = qb * 64 + wv * 16;
    const int hcol = h * HEAD_DIM;

    // Q fragments (16 rows x 64 hd -> two K=32 fragments), loaded once.
    v16bf qf[2];
    {
        const __bf16* qr = Q + (rowbase + q0 + mrow) * D_MODEL + hcol;
#pragma unroll
        for (int kk = 0; kk < 2; ++kk) {
            const int k = kk * 32 + ksel * 8;
            qf[kk] = load_frag(qr + k, qr + k + 16);
        }
    }

    v8f oacc[4];
#pragma unroll
    for (int nd = 0; nd < 4; ++nd) oacc[nd] = {};
    float mst[8], lst[8];
#pragma unroll
    for (int r = 0; r < 8; ++r) { mst[r] = -1e30f; lst[r] = 0.0f; }

    const int ldr = tid >> 1;   // 0..63 : staging row (key index)
    const int lsg = tid & 1;    // which 32-half segment of hd

    const __bf16* kgb = Kg + (rowbase + ldr) * D_MODEL + hcol + lsg * 32;
    const __bf16* vgb = V  + (rowbase + ldr) * D_MODEL + hcol + lsg * 32;
    uint32_t koff[2];
    koff[0] = lds_off(&Ks[0][ldr * KSTR + lsg * 32]);
    koff[1] = lds_off(&Ks[1][ldr * KSTR + lsg * 32]);

    __bf16 vtmp[32];

    // Prologue: stage tile 0 (K async, V via registers + transpose stores).
    {
        const uint64_t ga = (uint64_t)(uintptr_t)kgb;
        ASYNC_CP16(koff[0], ga, 0);  ASYNC_CP16(koff[0], ga, 16);
        ASYNC_CP16(koff[0], ga, 32); ASYNC_CP16(koff[0], ga, 48);
#pragma unroll
        for (int j = 0; j < 4; ++j)
            *(v8bf*)(vtmp + j * 8) = *(const v8bf*)(vgb + j * 8);
#pragma unroll
        for (int j = 0; j < 32; ++j)
            Vt[0][(lsg * 32 + j) * KSTR + ldr] = vtmp[j];
    }

    for (int kk0 = 0; kk0 < SEQ; kk0 += 64) {
        const int it = (kk0 >> 6) & 1;
        const bool more = (kk0 + 64) < SEQ;
        WAIT_ASYNC0();
        __syncthreads();

        if (more) {   // prefetch next K tile (async) and V tile (to VGPRs)
            const int nb = it ^ 1;
            const uint64_t ga =
                (uint64_t)(uintptr_t)(kgb + (size_t)(kk0 + 64) * D_MODEL);
            ASYNC_CP16(koff[nb], ga, 0);  ASYNC_CP16(koff[nb], ga, 16);
            ASYNC_CP16(koff[nb], ga, 32); ASYNC_CP16(koff[nb], ga, 48);
            const __bf16* vg = vgb + (size_t)(kk0 + 64) * D_MODEL;
#pragma unroll
            for (int j = 0; j < 4; ++j)
                *(v8bf*)(vtmp + j * 8) = *(const v8bf*)(vg + j * 8);
        }

        // S = Q @ K^T  (16 q-rows x 64 keys), K-dim = hd = 64.
        v8f sacc[4];
#pragma unroll
        for (int kn = 0; kn < 4; ++kn) sacc[kn] = {};
#pragma unroll
        for (int kk = 0; kk < 2; ++kk) {
            const int ko = kk * 32 + ksel * 8;
#pragma unroll
            for (int kn = 0; kn < 4; ++kn) {
                const int key = kn * 16 + mrow;
                v16bf bfr = load_frag(&Ks[it][key * KSTR + ko],
                                      &Ks[it][key * KSTR + ko + 16]);
                sacc[kn] = __builtin_amdgcn_wmma_f32_16x16x32_bf16(
                    false, qf[kk], false, bfr, (short)0, sacc[kn], false, false);
            }
        }

        // Gate + online-max rescale, write P (bf16) to per-wave LDS tile.
        __bf16* pw = Ps[wv];
#pragma unroll
        for (int r = 0; r < 8; ++r) {
            float mx = -1e30f;
#pragma unroll
            for (int kn = 0; kn < 4; ++kn) mx = fmaxf(mx, sacc[kn][r]);
            mx *= 0.125f;
            mx = rowmax16(mx);
            const float mnew  = fmaxf(mst[r], mx);
            const float alpha = __expf(mst[r] - mnew);
            mst[r] = mnew;
            float rsum = 0.0f;
#pragma unroll
            for (int kn = 0; kn < 4; ++kn) {
                const float sp = sacc[kn][r] * 0.125f;
                const float w  = __expf(sp - mnew) *
                                 __builtin_amdgcn_rcpf(1.0f + __expf(muv - sp));
                rsum += w;
                pw[(r + half * 8) * KSTR + kn * 16 + mrow] = (__bf16)w;
            }
            rsum = rowsum16(rsum);
            lst[r] = lst[r] * alpha + rsum;
#pragma unroll
            for (int nd = 0; nd < 4; ++nd) oacc[nd][r] *= alpha;
        }
        // Wave-local LDS RAW fence before re-reading P as A-fragments.
        WAIT_DS0();

        // O += P @ V  (K-dim = 64 keys).
#pragma unroll
        for (int kt = 0; kt < 2; ++kt) {
            const int ko = kt * 32 + ksel * 8;
            v16bf pf = load_frag(&pw[mrow * KSTR + ko],
                                 &pw[mrow * KSTR + ko + 16]);
#pragma unroll
            for (int nd = 0; nd < 4; ++nd) {
                const int d = nd * 16 + mrow;
                v16bf vf = load_frag(&Vt[it][d * KSTR + ko],
                                     &Vt[it][d * KSTR + ko + 16]);
                oacc[nd] = __builtin_amdgcn_wmma_f32_16x16x32_bf16(
                    false, pf, false, vf, (short)0, oacc[nd], false, false);
            }
        }

        if (more) {   // transpose-store prefetched V into the other buffer
            const int nb = it ^ 1;
#pragma unroll
            for (int j = 0; j < 32; ++j)
                Vt[nb][(lsg * 32 + j) * KSTR + ldr] = vtmp[j];
        }
    }

    // Normalize (v_rcp) and store bf16 attention output [b, s, h*hd].
    float inv[8];
#pragma unroll
    for (int r = 0; r < 8; ++r) inv[r] = __builtin_amdgcn_rcpf(lst[r]);
    __bf16* op = O + (rowbase + q0 + half * 8) * D_MODEL + hcol + mrow;
#pragma unroll
    for (int nd = 0; nd < 4; ++nd)
#pragma unroll
        for (int r = 0; r < 8; ++r)
            op[r * D_MODEL + nd * 16] = (__bf16)(oacc[nd][r] * inv[r]);
}

// ---------------------------------------------------------------------------
// Launch
// ---------------------------------------------------------------------------
extern "C" void kernel_launch(void* const* d_in, const int* in_sizes, int n_in,
                              void* d_out, int out_size, void* d_ws, size_t ws_size,
                              hipStream_t stream) {
    (void)in_sizes; (void)n_in; (void)out_size; (void)ws_size;

    const float* seq = (const float*)d_in[0];
    const float* gam = (const float*)d_in[1];
    const float* bet = (const float*)d_in[2];
    const float* Wq  = (const float*)d_in[3];
    const float* bq  = (const float*)d_in[4];
    const float* Wk  = (const float*)d_in[5];
    const float* bk  = (const float*)d_in[6];
    const float* Wv  = (const float*)d_in[7];
    const float* bv  = (const float*)d_in[8];
    const float* Wo  = (const float*)d_in[9];
    const float* bo  = (const float*)d_in[10];
    const float* mu  = (const float*)d_in[11];

    const int M = BATCH * SEQ;           // 8192

    char* ws = (char*)d_ws;
    const size_t actB = (size_t)M * D_MODEL * sizeof(__bf16);       // 16 MiB
    const size_t wB   = (size_t)D_MODEL * D_MODEL * sizeof(__bf16); //  2 MiB
    __bf16* xb    = (__bf16*)(ws);
    __bf16* wqb   = (__bf16*)(ws + actB);
    __bf16* wkb   = (__bf16*)(ws + actB + wB);
    __bf16* wvb   = (__bf16*)(ws + actB + 2 * wB);
    __bf16* wob   = (__bf16*)(ws + actB + 3 * wB);
    __bf16* Qb    = (__bf16*)(ws + actB + 4 * wB);
    __bf16* Kb    = (__bf16*)(ws + 2 * actB + 4 * wB);
    __bf16* Vb    = (__bf16*)(ws + 3 * actB + 4 * wB);
    __bf16* attnb = (__bf16*)(ws + 4 * actB + 4 * wB);

    // 1) LayerNorm -> bf16 activations
    ln_bf16_kernel<<<M, 256, 0, stream>>>(seq, gam, bet, xb);

    // 2) Weights -> bf16
    const int WN = D_MODEL * D_MODEL;
    f32_to_bf16_kernel<<<1024, 256, 0, stream>>>(Wq, wqb, WN);
    f32_to_bf16_kernel<<<1024, 256, 0, stream>>>(Wk, wkb, WN);
    f32_to_bf16_kernel<<<1024, 256, 0, stream>>>(Wv, wvb, WN);
    f32_to_bf16_kernel<<<1024, 256, 0, stream>>>(Wo, wob, WN);

    // 3) Q/K/V projections (bf16 out)
    dim3 gg(GN / 128, M / 128);
    gemm_bf16_kernel<false><<<gg, 256, 0, stream>>>(xb, wqb, bq, Qb);
    gemm_bf16_kernel<false><<<gg, 256, 0, stream>>>(xb, wkb, bk, Kb);
    gemm_bf16_kernel<false><<<gg, 256, 0, stream>>>(xb, wvb, bv, Vb);

    // 4) Attention
    attn_kernel<<<BATCH * N_HEADS * (SEQ / 64), 128, 0, stream>>>(
        Qb, Kb, Vb, mu, attnb);

    // 5) Output projection (fp32 out)
    gemm_bf16_kernel<true><<<gg, 256, 0, stream>>>(attnb, wob, bo, d_out);
}
